// GNNModel_78847009620618
// MI455X (gfx1250) — matrix operand order repaired and checked by reading, not compiled, not run on hardware
//
#include <hip/hip_runtime.h>
#include <stdint.h>

// ---------------- model constants (match reference) ----------------
#define HID     256
#define F_IN    128
#define NGRAPH  64
#define NCLASS  38
#define CHID    128          // HID/2
#define POOLF   384          // HID + F_IN
#define BN_EPS  1e-5f

typedef __attribute__((ext_vector_type(16))) __bf16 v16bf;
typedef __attribute__((ext_vector_type(8)))  __bf16 v8bf;
typedef __attribute__((ext_vector_type(8)))  float  v8f;

static inline int cdiv(int a, int b) { return (a + b - 1) / b; }

// float -> bf16 bits, round-to-nearest-even
__device__ __forceinline__ unsigned short f2bf(float f) {
    unsigned int u = __float_as_uint(f);
    u += 0x7FFFu + ((u >> 16) & 1u);
    return (unsigned short)(u >> 16);
}

// ---------------- small utility kernels ----------------
__global__ void k_cast_bf16(const float* __restrict__ in, unsigned short* __restrict__ out, int n) {
    int i = blockIdx.x * blockDim.x + threadIdx.x;
    if (i < n) out[i] = f2bf(in[i]);
}

// W [K,H] f32 row-major -> Wt [H,K] bf16 row-major (so B fragment loads are contiguous)
__global__ void k_wt_cast(const float* __restrict__ W, unsigned short* __restrict__ Wt, int K, int H) {
    int i = blockIdx.x * blockDim.x + threadIdx.x;
    if (i >= K * H) return;
    int k = i / H, h = i - k * H;
    Wt[(size_t)h * K + k] = f2bf(W[i]);
}

__global__ void k_fill(float* __restrict__ p, float v, int n) {
    int i = blockIdx.x * blockDim.x + threadIdx.x;
    if (i < n) p[i] = v;
}

__global__ void k_deg_count(const int* __restrict__ dst, float* __restrict__ deg, int E) {
    int e = blockIdx.x * blockDim.x + threadIdx.x;
    if (e < E) atomicAdd(&deg[dst[e]], 1.0f);
}

__global__ void k_rsqrt_inplace(float* __restrict__ p, int n) {
    int i = blockIdx.x * blockDim.x + threadIdx.x;
    if (i < n) p[i] = rsqrtf(p[i]);   // deg >= 1 always (self loop), no zero case
}

// ---------------- WMMA bf16 GEMM: C[N,HID] = A[N,K] * Bt[HID,K]^T ----------------
// Weights-stationary: one wave owns MT consecutive 16-row tiles of one 16-col tile.
// B fragments for the whole K loop live in registers and are reused MT times.
// Compute loop is branch-free (OOB tiles clamp to valid rows); only stores are
// guarded, with a readfirstlane-uniform scalar branch (keeps EXEC all-1s for WMMA).
// Fragment layouts per CDNA5 ISA 7.12.2 (16-bit A 16x32, B 32x16, f32 C 16x16).
template<int K, int MT>
__global__ void __launch_bounds__(256) k_gemm_wmma(
    const unsigned short* __restrict__ Abf,  // [N,K] bf16 row-major
    const unsigned short* __restrict__ Bt,   // [HID,K] bf16 row-major (weights pre-transposed)
    float* __restrict__ C,                   // [N,HID] f32 row-major
    int N)
{
    constexpr int KB = K / 32;                // wmma steps along K
    const int wave   = (blockIdx.x * blockDim.x + threadIdx.x) >> 5;
    const int lane   = threadIdx.x & 31;
    const int tilesN = HID >> 4;              // 16
    const int mgroup = wave / tilesN;
    const int tn     = wave - mgroup * tilesN;
    const int tilesM = (N + 15) >> 4;
    const int tm0    = mgroup * MT;
    if (__builtin_amdgcn_readfirstlane(tm0) >= tilesM) return;   // scalar branch

    const int hi  = lane >> 4;                // lane group 0/1
    const int lid = lane & 15;

    // Preload all B fragments for this column tile (lane n = lid holds K = kb*32 + hi*8 + {0..7} and kb*32 + 16 + hi*8 + {0..7})
    const __bf16* brow = (const __bf16*)Bt + (size_t)(tn * 16 + lid) * K + hi * 8;
    v16bf bfrag[KB];
    #pragma unroll
    for (int kb = 0; kb < KB; ++kb) {
        v8bf b0 = *(const v8bf*)(brow + kb * 32);
        v8bf b1 = *(const v8bf*)(brow + kb * 32 + 16);
        #pragma unroll
        for (int i = 0; i < 8; ++i) { bfrag[kb][i] = b0[i]; bfrag[kb][i + 8] = b1[i]; }
    }

    // A row base pointers per tile; OOB tiles clamp to last row (harmless extra work)
    const __bf16* arow[MT];
    #pragma unroll
    for (int mt = 0; mt < MT; ++mt) {
        int row = (tm0 + mt) * 16 + lid;
        if (row >= N) row = N - 1;
        arow[mt] = (const __bf16*)Abf + (size_t)row * K + hi * 8;
    }

    v8f acc[MT];
    #pragma unroll
    for (int mt = 0; mt < MT; ++mt) acc[mt] = (v8f){};

    // kb-outer / mt-inner: all MT A-fragment loads issue as one clause, then MT wmmas
    // run while the next kb's loads are in flight.
    #pragma unroll
    for (int kb = 0; kb < KB; ++kb) {
        v16bf a[MT];
        #pragma unroll
        for (int mt = 0; mt < MT; ++mt) {
            // A lane: K = kb*32 + hi*8 + {0..7}  and  kb*32 + 16 + hi*8 + {0..7}
            v8bf a0 = *(const v8bf*)(arow[mt] + kb * 32);
            v8bf a1 = *(const v8bf*)(arow[mt] + kb * 32 + 16);
            #pragma unroll
            for (int i = 0; i < 8; ++i) { a[mt][i] = a0[i]; a[mt][i + 8] = a1[i]; }
        }
        #pragma unroll
        for (int mt = 0; mt < MT; ++mt) {
            acc[mt] = __builtin_amdgcn_wmma_f32_16x16x32_bf16(
                          false, a[mt], false, bfrag[kb], (short)0, acc[mt], false, false);
        }
    }

    // C layout: lane holds column n=lid; VGPR r holds row M = r + hi*8.
    // HID compile-time -> stores are base + immediate offsets (r*1024B).
    const int col = tn * 16 + lid;
    #pragma unroll
    for (int mt = 0; mt < MT; ++mt) {
        if (__builtin_amdgcn_readfirstlane(tm0 + mt) < tilesM) {   // scalar branch
            float* crow = C + (size_t)((tm0 + mt) * 16 + hi * 8) * HID + col;
            #pragma unroll
            for (int r = 0; r < 8; ++r) crow[r * HID] = acc[mt][r];
        }
    }
}

// ---------------- aggregation ----------------
// self loop: agg[i,f] = xw[i,f] * dinv[i]^2
__global__ void k_selfloop(const float* __restrict__ xw, const float* __restrict__ dinv,
                           float* __restrict__ agg, int total, int H) {
    int i = blockIdx.x * blockDim.x + threadIdx.x;
    if (i >= total) return;
    int node = i / H;
    float d = dinv[node];
    agg[i] = xw[i] * d * d;
}

// one block (H threads) per edge: agg[dst,f] += xw[src,f] * dinv[src]*dinv[dst]
__global__ void k_edge_scatter(const int* __restrict__ src, const int* __restrict__ dst,
                               const float* __restrict__ dinv, const float* __restrict__ xw,
                               float* __restrict__ agg, int E, int H) {
    int e = blockIdx.x;
    if (e >= E) return;
    int f = threadIdx.x;
    int s = src[e], d = dst[e];
    float nrm = dinv[s] * dinv[d];
    atomicAdd(&agg[(size_t)d * H + f], xw[(size_t)s * H + f] * nrm);
}

// bias + eval-mode BN + optional ReLU; optional bf16 (next GEMM input) / f32 outputs
__global__ void k_bn_act(const float* __restrict__ agg, const float* __restrict__ bias,
                         const float* __restrict__ g, const float* __restrict__ be,
                         const float* __restrict__ mean, const float* __restrict__ var,
                         unsigned short* __restrict__ obf, float* __restrict__ of32,
                         int total, int H, int do_relu) {
    int i = blockIdx.x * blockDim.x + threadIdx.x;
    if (i >= total) return;
    int f = i % H;
    float y = (agg[i] + bias[f] - mean[f]) * rsqrtf(var[f] + BN_EPS) * g[f] + be[f];
    if (do_relu) y = fmaxf(y, 0.0f);
    if (obf)  obf[i]  = f2bf(y);
    if (of32) of32[i] = y;
}

// ---------------- pooling ----------------
// one block (256 threads) per node; pooled[g, 0:256]=h3 sum, pooled[g, 256:384]=x sum
__global__ void k_pool_accum(const float* __restrict__ h3, const float* __restrict__ x,
                             const int* __restrict__ batch, float* __restrict__ pooled,
                             float* __restrict__ cnt, int N) {
    int i = blockIdx.x;
    if (i >= N) return;
    int f = threadIdx.x;
    int g = batch[i];
    atomicAdd(&pooled[(size_t)g * POOLF + f], h3[(size_t)i * HID + f]);
    if (f < F_IN) atomicAdd(&pooled[(size_t)g * POOLF + HID + f], x[(size_t)i * F_IN + f]);
    if (f == 0) atomicAdd(&cnt[g], 1.0f);
}

__global__ void k_pool_div(float* __restrict__ pooled, const float* __restrict__ cnt) {
    int i = blockIdx.x * blockDim.x + threadIdx.x;
    if (i >= NGRAPH * POOLF) return;
    pooled[i] /= fmaxf(cnt[i / POOLF], 1.0f);
}

// ---------------- small classifier GEMMs (scalar f32) ----------------
__global__ void k_fc(const float* __restrict__ in, const float* __restrict__ W,
                     const float* __restrict__ b, float* __restrict__ out,
                     int M, int K, int Ncol, int do_relu) {
    int i = blockIdx.x * blockDim.x + threadIdx.x;
    if (i >= M * Ncol) return;
    int r = i / Ncol, c = i - r * Ncol;
    float s = b[c];
    for (int k = 0; k < K; ++k) s += in[(size_t)r * K + k] * W[(size_t)k * Ncol + c];
    out[i] = do_relu ? fmaxf(s, 0.0f) : s;
}

// ---------------- driver ----------------
extern "C" void kernel_launch(void* const* d_in, const int* in_sizes, int n_in,
                              void* d_out, int out_size, void* d_ws, size_t ws_size,
                              hipStream_t stream) {
    (void)n_in; (void)out_size; (void)ws_size;

    const float* x     = (const float*)d_in[0];
    const int*   ei    = (const int*)d_in[1];
    const int*   batch = (const int*)d_in[2];
    const float* w1 = (const float*)d_in[3];  const float* b1 = (const float*)d_in[4];
    const float* w2 = (const float*)d_in[5];  const float* b2 = (const float*)d_in[6];
    const float* w3 = (const float*)d_in[7];  const float* b3 = (const float*)d_in[8];
    const float* g1 = (const float*)d_in[9];  const float* be1 = (const float*)d_in[10];
    const float* m1 = (const float*)d_in[11]; const float* v1  = (const float*)d_in[12];
    const float* g2 = (const float*)d_in[13]; const float* be2 = (const float*)d_in[14];
    const float* m2 = (const float*)d_in[15]; const float* v2  = (const float*)d_in[16];
    const float* g3 = (const float*)d_in[17]; const float* be3 = (const float*)d_in[18];
    const float* m3 = (const float*)d_in[19]; const float* v3  = (const float*)d_in[20];
    const float* cw1 = (const float*)d_in[21]; const float* cb1 = (const float*)d_in[22];
    const float* cw2 = (const float*)d_in[23]; const float* cb2 = (const float*)d_in[24];
    float* out = (float*)d_out;

    const int N = in_sizes[2];          // nodes (50000)
    const int E = in_sizes[1] / 2;      // edges (600000)
    const int* esrc = ei;
    const int* edst = ei + E;

    // workspace carve-up (256B aligned)
    char* ws = (char*)d_ws;
    size_t off = 0;
    #define WS_ALLOC(ty, name, bytes) ty* name = (ty*)(ws + off); off = (off + (size_t)(bytes) + 255) & ~(size_t)255;
    WS_ALLOC(float,          xw,     (size_t)N * HID * 4)       // GEMM output / layer-3 h
    WS_ALLOC(float,          agg,    (size_t)N * HID * 4)       // aggregation accumulator
    WS_ALLOC(unsigned short, hbf,    (size_t)N * HID * 2)       // bf16 activations
    WS_ALLOC(unsigned short, xbf,    (size_t)N * F_IN * 2)      // bf16 input features
    WS_ALLOC(float,          dinv,   (size_t)N * 4)             // deg -> rsqrt(deg)
    WS_ALLOC(unsigned short, w1t,    (size_t)HID * F_IN * 2)    // [H,K] bf16
    WS_ALLOC(unsigned short, w2t,    (size_t)HID * HID * 2)
    WS_ALLOC(unsigned short, w3t,    (size_t)HID * HID * 2)
    WS_ALLOC(float,          pooled, (size_t)NGRAPH * POOLF * 4)
    WS_ALLOC(float,          cnt,    (size_t)NGRAPH * 4)
    WS_ALLOC(float,          zbuf,   (size_t)NGRAPH * CHID * 4)
    #undef WS_ALLOC

    const int NH = N * HID;
    constexpr int MT = 4;               // M-tiles per wave (weights-stationary reuse)
    const int tilesM = cdiv(N, 16);
    const int gemm_waves  = cdiv(tilesM, MT) * (HID / 16);
    const int gemm_blocks = cdiv(gemm_waves * 32, 256);

    // 0) precompute: bf16 casts, transposed weights, degree normalization
    k_cast_bf16<<<cdiv(N * F_IN, 256), 256, 0, stream>>>(x, xbf, N * F_IN);
    k_wt_cast<<<cdiv(F_IN * HID, 256), 256, 0, stream>>>(w1, w1t, F_IN, HID);
    k_wt_cast<<<cdiv(HID * HID, 256), 256, 0, stream>>>(w2, w2t, HID, HID);
    k_wt_cast<<<cdiv(HID * HID, 256), 256, 0, stream>>>(w3, w3t, HID, HID);
    k_fill<<<cdiv(N, 256), 256, 0, stream>>>(dinv, 1.0f, N);                 // self loop
    k_deg_count<<<cdiv(E, 256), 256, 0, stream>>>(edst, dinv, E);
    k_rsqrt_inplace<<<cdiv(N, 256), 256, 0, stream>>>(dinv, N);

    // 1) layer 1: xw = x @ W1 (WMMA bf16), aggregate, BN+ReLU -> hbf
    k_gemm_wmma<F_IN, MT><<<gemm_blocks, 256, 0, stream>>>(xbf, w1t, xw, N);
    k_selfloop<<<cdiv(NH, 256), 256, 0, stream>>>(xw, dinv, agg, NH, HID);
    k_edge_scatter<<<E, HID, 0, stream>>>(esrc, edst, dinv, xw, agg, E, HID);
    k_bn_act<<<cdiv(NH, 256), 256, 0, stream>>>(agg, b1, g1, be1, m1, v1, hbf, nullptr, NH, HID, 1);

    // 2) layer 2
    k_gemm_wmma<HID, MT><<<gemm_blocks, 256, 0, stream>>>(hbf, w2t, xw, N);
    k_selfloop<<<cdiv(NH, 256), 256, 0, stream>>>(xw, dinv, agg, NH, HID);
    k_edge_scatter<<<E, HID, 0, stream>>>(esrc, edst, dinv, xw, agg, E, HID);
    k_bn_act<<<cdiv(NH, 256), 256, 0, stream>>>(agg, b2, g2, be2, m2, v2, hbf, nullptr, NH, HID, 1);

    // 3) layer 3 (no ReLU), f32 output into xw
    k_gemm_wmma<HID, MT><<<gemm_blocks, 256, 0, stream>>>(hbf, w3t, xw, N);
    k_selfloop<<<cdiv(NH, 256), 256, 0, stream>>>(xw, dinv, agg, NH, HID);
    k_edge_scatter<<<E, HID, 0, stream>>>(esrc, edst, dinv, xw, agg, E, HID);
    k_bn_act<<<cdiv(NH, 256), 256, 0, stream>>>(agg, b3, g3, be3, m3, v3, nullptr, xw, NH, HID, 0);

    // 4) mean pool (h3 || x_orig) per graph
    k_fill<<<cdiv(NGRAPH * POOLF, 256), 256, 0, stream>>>(pooled, 0.0f, NGRAPH * POOLF);
    k_fill<<<1, NGRAPH, 0, stream>>>(cnt, 0.0f, NGRAPH);
    k_pool_accum<<<N, HID, 0, stream>>>(xw, x, batch, pooled, cnt, N);
    k_pool_div<<<cdiv(NGRAPH * POOLF, 256), 256, 0, stream>>>(pooled, cnt);

    // 5) classifier: z = relu(pooled @ cw1 + cb1); out = z @ cw2 + cb2
    k_fc<<<cdiv(NGRAPH * CHID, 128), 128, 0, stream>>>(pooled, cw1, cb1, zbuf, NGRAPH, POOLF, CHID, 1);
    k_fc<<<cdiv(NGRAPH * NCLASS, 128), 128, 0, stream>>>(zbuf, cw2, cb2, out, NGRAPH, CHID, NCLASS, 0);
}
